// MultiHeadCrossAttention_30150670418086
// MI455X (gfx1250) — matrix operand order repaired and checked by reading, not compiled, not run on hardware
//
#include <hip/hip_runtime.h>

typedef __attribute__((ext_vector_type(16))) _Float16 v16h;
typedef __attribute__((ext_vector_type(8)))  float    v8f;
typedef __attribute__((ext_vector_type(4)))  float    v4f;
typedef __attribute__((ext_vector_type(4)))  _Float16 v4h;

#define BB  16
#define SS  512
#define HH  16
#define DD  64
#define HID 1024
#define MM  (BB*SS)   // 8192

// ---------------- CDNA5 async copy: global -> LDS, 128-bit, ASYNCcnt tracked ----------------

__device__ inline void async_ld_b128(void* lds, const void* g) {
  asm volatile("global_load_async_to_lds_b128 %0, %1, off"
               :: "v"((unsigned)(unsigned long long)lds), "v"(g)
               : "memory");
}
template<int N>
__device__ inline void wait_async() {
  asm volatile("s_wait_asynccnt %0" :: "i"(N) : "memory");
}

// ---------------- WMMA helpers (CDNA5 16x16x32 f16, f32 accum) ----------------

__device__ inline v8f wmma_f16(v16h a, v16h b, v8f c) {
  return __builtin_amdgcn_wmma_f32_16x16x32_f16(false, a, false, b, (short)0, c, false, false);
}

// A matrix 16x32 f16, row-major LDS tile (ISA 7.12.2 lane layout)
__device__ inline v16h load_a(const _Float16* p, int ld) {
  int lane = threadIdx.x & 31;
  int m  = lane & 15;
  int kb = (lane >> 4) << 3;
  const _Float16* row = p + m * ld;
  v16h a;
#pragma unroll
  for (int i = 0; i < 4; ++i) {
    a[2*i]     = row[kb + 2*i];
    a[2*i + 1] = row[kb + 2*i + 1];
    a[8 + 2*i] = row[16 + kb + 2*i];
    a[9 + 2*i] = row[16 + kb + 2*i + 1];
  }
  return a;
}

// B matrix 32x16 f16 from K-major LDS tile: lane%16 = N col, elts = K rows (table order)
__device__ inline v16h load_b(const _Float16* p, int ld) {
  int lane = threadIdx.x & 31;
  int n  = lane & 15;
  int kb = (lane >> 4) << 3;
  v16h b;
#pragma unroll
  for (int j = 0; j < 8; ++j) {
    b[j]     = p[(kb + j) * ld + n];
    b[8 + j] = p[(16 + kb + j) * ld + n];
  }
  return b;
}

// ---------------- Prep kernels: one-time f32 -> f16 conversions ----------------

__global__ __launch_bounds__(256) void cvt16(const float* __restrict__ s,
                                             _Float16* __restrict__ d, int n4) {
  int i = blockIdx.x * 256 + threadIdx.x;
  if (i < n4) {
    v4f x = ((const v4f*)s)[i];
    v4h y;
#pragma unroll
    for (int j = 0; j < 4; ++j) y[j] = (_Float16)x[j];
    ((v4h*)d)[i] = y;
  }
}

// W [HID x K] f32 -> WT [K x HID] f16
__global__ __launch_bounds__(256) void cvtWT(const float* __restrict__ W,
                                             _Float16* __restrict__ WT, int K) {
  int i = blockIdx.x * 256 + threadIdx.x;
  if (i < K * HID) {
    int k = i >> 10, n = i & 1023;
    WT[i] = (_Float16)W[(size_t)n * K + k];
  }
}

// attention weights -> f16, transposed to B-matrix layout:
// dst[((dir*HH+h)*4 + m)*4096 + d*64 + e] = W_m[e][d]   (m: 0=q,1=k,2=v,3=o)
__global__ __launch_bounds__(256) void cvt_attn_w(const float* __restrict__ iw,
                                                  const float* __restrict__ ow,
                                                  const float* __restrict__ tw,
                                                  const float* __restrict__ tow,
                                                  _Float16* __restrict__ dst) {
  int i = blockIdx.x * 256 + threadIdx.x;          // 0 .. 2*16*4*64*64-1
  int dir = i >> 18;
  int r   = i & ((1 << 18) - 1);
  int h   = r >> 14;
  int m   = (r >> 12) & 3;
  int d   = (r >> 6) & 63;
  int e   = r & 63;
  const float* inw  = dir ? tw  : iw;
  const float* outw = dir ? tow : ow;
  float v = (m < 3) ? inw[((size_t)h * 192 + m * 64 + e) * 64 + d]
                    : outw[((size_t)h * 64 + e) * 64 + d];
  dst[i] = (_Float16)v;
}

// ---------------- Kernel 1a: Y = X16 @ WT16 + bias  (WMMA GEMM, double-buffered async) ------
// X16: [MM x K] f16, WT16: [K x HID] f16, Y: [MM x HID] f32. Tile 128x128, block 256.

#define GEMM_BUF (128*40 + 32*136)   // halves per (A+B) buffer = 9472

__global__ __launch_bounds__(256) void proj_gemm(const _Float16* __restrict__ X16,
                                                 const _Float16* __restrict__ WT16,
                                                 const float* __restrict__ bias,
                                                 float* __restrict__ Y, int K) {
  extern __shared__ char smem[];
  const int LDA = 40, LDB = 136;
  _Float16* base = (_Float16*)smem;

  int m0 = blockIdx.x * 128;
  int n0 = blockIdx.y * 128;
  int tid  = threadIdx.x;
  int wave = tid >> 5, lane = tid & 31;
  int nl = lane & 15, mb = (lane >> 4) * 8;

  // issue one K-chunk's tiles into buffer `buf` (4 async instructions per wave)
  auto issue = [&](int buf, int kk) {
    _Float16* aT = base + buf * GEMM_BUF;
    _Float16* bT = aT + 128 * LDA;
#pragma unroll
    for (int c = tid; c < 512; c += 256) {          // A tile 128x32
      int r = c >> 2, q = c & 3;
      async_ld_b128(aT + r * LDA + q * 8, X16 + (size_t)(m0 + r) * K + kk + q * 8);
    }
#pragma unroll
    for (int c = tid; c < 512; c += 256) {          // B tile 32x128
      int k = c >> 4, q = c & 15;
      async_ld_b128(bT + k * LDB + q * 8, WT16 + (size_t)(kk + k) * HID + n0 + q * 8);
    }
  };

  v8f acc[8] = {};
  int nc = K >> 5;
  issue(0, 0);

  for (int ic = 0; ic < nc; ++ic) {
    int cur = ic & 1;
    if (ic + 1 < nc) { issue(cur ^ 1, (ic + 1) << 5); wait_async<4>(); }
    else             { wait_async<0>(); }
    __syncthreads();                                 // tiles ready for all waves

    _Float16* aT = base + cur * GEMM_BUF;
    _Float16* bT = aT + 128 * LDA;
    v16h a = load_a(aT + wave * 16 * LDA, LDA);
    v16h bf[8];
#pragma unroll
    for (int t = 0; t < 8; ++t) bf[t] = load_b(bT + t * 16, LDB);
#pragma unroll
    for (int t = 0; t < 8; ++t) acc[t] = wmma_f16(a, bf[t], acc[t]);
    __syncthreads();                                 // readers done before overwrite
  }
#pragma unroll
  for (int t = 0; t < 8; ++t) {
    int n = n0 + t * 16 + nl;
    float bv = bias[n];
#pragma unroll
    for (int i = 0; i < 8; ++i) {
      int m = m0 + wave * 16 + mb + i;
      Y[(size_t)m * HID + n] = acc[t][i] + bv;
    }
  }
}

// ---------------- Kernel 1b: LayerNorm + L2-normalize + D^-0.5, reshape to (b,h,s,d) f16 ----

__global__ __launch_bounds__(256) void ln_l2_reshape(const float* __restrict__ Y,
                                                     const float* __restrict__ g,
                                                     const float* __restrict__ be,
                                                     _Float16* __restrict__ outH) {
  int row = blockIdx.x;              // b*SS + s
  int b = row >> 9, s = row & 511;
  const float* x = Y + (size_t)row * HID;
  int tid = threadIdx.x, wave = tid >> 5, lane = tid & 31;

  __shared__ float red[32];
  float v[4], sum = 0.f, sq = 0.f;
#pragma unroll
  for (int i = 0; i < 4; ++i) { v[i] = x[tid + i * 256]; sum += v[i]; sq += v[i] * v[i]; }
  for (int off = 16; off >= 1; off >>= 1) { sum += __shfl_xor(sum, off, 32); sq += __shfl_xor(sq, off, 32); }
  if (lane == 0) { red[wave] = sum; red[8 + wave] = sq; }
  __syncthreads();
  sum = 0.f; sq = 0.f;
#pragma unroll
  for (int w = 0; w < 8; ++w) { sum += red[w]; sq += red[8 + w]; }
  float mu   = sum * (1.f / HID);
  float var  = sq * (1.f / HID) - mu * mu;
  float rstd = rsqrtf(var + 1e-5f);

  float u[4], nrm = 0.f;
#pragma unroll
  for (int i = 0; i < 4; ++i) {
    int j = tid + i * 256;
    u[i] = (v[i] - mu) * rstd * g[j] + be[j];
    nrm += u[i] * u[i];
  }
  for (int off = 16; off >= 1; off >>= 1) nrm += __shfl_xor(nrm, off, 32);
  if (lane == 0) red[16 + wave] = nrm;
  __syncthreads();
  nrm = 0.f;
#pragma unroll
  for (int w = 0; w < 8; ++w) nrm += red[16 + w];
  float inv = 0.125f / fmaxf(sqrtf(nrm), 1e-12f);   // L2-normalize * D^-0.5
#pragma unroll
  for (int i = 0; i < 4; ++i) {
    int j = tid + i * 256;
    int h = j >> 6, d = j & 63;
    outH[(((size_t)b * HH + h) * SS + s) * DD + d] = (_Float16)(u[i] * inv);
  }
}

// ---------------- Kernel 2: fused per-head cross attention (flash, pipelined KV DMA) --------
// grid: (8 = 2 dirs * 4 q-tiles, H, B); block 256 (8 waves, one 16-row q-slab per wave).

__global__ __launch_bounds__(256) void cross_attn_kernel(
    const _Float16* __restrict__ ciH, const _Float16* __restrict__ ctH,
    const _Float16* __restrict__ wT16,        // [2][16][4][64][64] f16, B-layout
    const float* __restrict__ i2t_in_b, const float* __restrict__ i2t_out_b,
    const float* __restrict__ t2i_in_b, const float* __restrict__ t2i_out_b,
    float* __restrict__ oI, float* __restrict__ oT) {
  int dir = blockIdx.x & 1;
  int qt  = blockIdx.x >> 1;       // 0..3
  int h = blockIdx.y, b = blockIdx.z;

  const _Float16* Xq   = (dir == 0 ? ciH : ctH) + ((size_t)b * HH + h) * SS * DD;
  const _Float16* Xkv  = (dir == 0 ? ctH : ciH) + ((size_t)b * HH + h) * SS * DD;
  const _Float16* wsrc = wT16 + ((size_t)(dir * HH + h)) * 4 * 4096;
  const float* in_b  = (dir == 0 ? i2t_in_b  : t2i_in_b)  + h * 192;
  const float* out_b = (dir == 0 ? i2t_out_b : t2i_out_b) + h * 64;
  float* O = (dir == 0 ? oI : oT) + ((size_t)b * HH + h) * SS * DD;

  extern __shared__ char smem[];
  const int LDW = 72, LDK = 136;
  _Float16* wq    = (_Float16*)smem;        // [4][64][72]: q,k,v,o (B layout [k][n])
  _Float16* wk    = wq + 64 * LDW;
  _Float16* wv    = wk + 64 * LDW;
  _Float16* wo    = wv + 64 * LDW;
  _Float16* stA   = wo + 64 * LDW;          // [128][72] stage buffer A
  _Float16* stB   = stA + 128 * LDW;        // [128][72] stage buffer B
  _Float16* qs    = stB + 128 * LDW;        // [128][72] Q (scaled, +bias)
  _Float16* kT    = qs + 128 * LDW;         // [64][136] K^T tile
  _Float16* vs    = kT + 64 * LDK;          // [128][72] V tile
  _Float16* ps    = vs + 128 * LDW;         // per-wave [16][136] P / O staging
  float* bq = (float*)(ps + 8 * 16 * LDK);
  float* bk = bq + 64; float* bv = bk + 64; float* bo = bv + 64;

  int tid = threadIdx.x, wave = tid >> 5, lane = tid & 31;
  int nl = lane & 15, mb = (lane >> 4) * 8;
  _Float16* pw = ps + wave * 16 * LDK;
  _Float16* kvbuf[2] = { stB, stA };        // KV tile jt lives in kvbuf[jt&1]

  auto issue_kv = [&](int jt, _Float16* dst) {   // 4 async instructions per wave
#pragma unroll
    for (int c = tid; c < 1024; c += 256) {
      int r = c >> 3, q = c & 7;
      async_ld_b128(dst + r * LDW + q * 8, Xkv + (size_t)(jt * 128 + r) * DD + q * 8);
    }
  };

  // prologue: weights (8/wave) + Q tile (4/wave) + KV tile 0 (4/wave)
#pragma unroll
  for (int c = tid; c < 2048; c += 256) {
    int row = c >> 3, q = c & 7;
    async_ld_b128(wq + row * LDW + q * 8, wsrc + (size_t)row * 64 + q * 8);
  }
#pragma unroll
  for (int c = tid; c < 1024; c += 256) {
    int r = c >> 3, q = c & 7;
    async_ld_b128(stA + r * LDW + q * 8, Xq + (size_t)(qt * 128 + r) * DD + q * 8);
  }
  issue_kv(0, stB);
  if (tid < 64) { bq[tid] = in_b[tid]; bk[tid] = in_b[64 + tid];
                  bv[tid] = in_b[128 + tid]; bo[tid] = out_b[tid]; }
  wait_async<4>();                 // weights + Q done (in-order); KV0 may be in flight
  __syncthreads();

  // Q = (Xq @ Wq^T + bq) * D^-0.5
  {
    v16h a0 = load_a(stA + wave * 16 * LDW, LDW);
    v16h a1 = load_a(stA + wave * 16 * LDW + 32, LDW);
#pragma unroll
    for (int t = 0; t < 4; ++t) {
      v8f c = {};
      c = wmma_f16(a0, load_b(wq + t * 16, LDW), c);
      c = wmma_f16(a1, load_b(wq + 32 * LDW + t * 16, LDW), c);
      int e = t * 16 + nl;
      float bb = bq[e];
#pragma unroll
      for (int i = 0; i < 8; ++i)
        qs[(wave * 16 + mb + i) * LDW + e] = (_Float16)((c[i] + bb) * 0.125f);
    }
  }
  __syncthreads();                 // all waves done reading stA before KV1 DMA targets it

  float mrow[8], lrow[8];
  v8f oacc[4] = {};
#pragma unroll
  for (int i = 0; i < 8; ++i) { mrow[i] = -1e30f; lrow[i] = 0.f; }

  for (int jt = 0; jt < 4; ++jt) {
    _Float16* cur = kvbuf[jt & 1];
    if (jt < 3) { issue_kv(jt + 1, kvbuf[(jt + 1) & 1]); wait_async<4>(); }
    else        { wait_async<0>(); }
    __syncthreads();               // KV tile ready; PV(jt-1) readers of vs all done

    // K, V projections for this kv tile (each wave: its 16 rows)
    {
      v16h a0 = load_a(cur + wave * 16 * LDW, LDW);
      v16h a1 = load_a(cur + wave * 16 * LDW + 32, LDW);
#pragma unroll
      for (int t = 0; t < 4; ++t) {
        int e = t * 16 + nl;
        v8f ck = {};
        ck = wmma_f16(a0, load_b(wk + t * 16, LDW), ck);
        ck = wmma_f16(a1, load_b(wk + 32 * LDW + t * 16, LDW), ck);
        float bkv = bk[e];
#pragma unroll
        for (int i = 0; i < 8; ++i)                // transposed store: kT[d][n]
          kT[e * LDK + wave * 16 + mb + i] = (_Float16)(ck[i] + bkv);
        v8f cv = {};
        cv = wmma_f16(a0, load_b(wv + t * 16, LDW), cv);
        cv = wmma_f16(a1, load_b(wv + 32 * LDW + t * 16, LDW), cv);
        float bvv = bv[e];
#pragma unroll
        for (int i = 0; i < 8; ++i)
          vs[(wave * 16 + mb + i) * LDW + e] = (_Float16)(cv[i] + bvv);
      }
    }
    __syncthreads();               // kT/vs visible to all waves

    // S = Q(16x64) @ K^T(64x128)
    v16h qa0 = load_a(qs + wave * 16 * LDW, LDW);
    v16h qa1 = load_a(qs + wave * 16 * LDW + 32, LDW);
    v8f sc[8];
#pragma unroll
    for (int t = 0; t < 8; ++t) {
      v8f c = {};
      c = wmma_f16(qa0, load_b(kT + t * 16, LDK), c);
      c = wmma_f16(qa1, load_b(kT + 32 * LDK + t * 16, LDK), c);
      sc[t] = c;
    }

    // online softmax (rows live on 16-lane groups; reduce with shfl_xor 1,2,4,8)
#pragma unroll
    for (int i = 0; i < 8; ++i) {
      float mx = sc[0][i];
#pragma unroll
      for (int t = 1; t < 8; ++t) mx = fmaxf(mx, sc[t][i]);
      for (int off = 1; off < 16; off <<= 1) mx = fmaxf(mx, __shfl_xor(mx, off, 32));
      float mnew = fmaxf(mrow[i], mx);
      float fac  = __expf(mrow[i] - mnew);
      mrow[i] = mnew;
      float ls = 0.f;
#pragma unroll
      for (int t = 0; t < 8; ++t) {
        float p = __expf(sc[t][i] - mnew);
        sc[t][i] = p;
        ls += p;
      }
      for (int off = 1; off < 16; off <<= 1) ls += __shfl_xor(ls, off, 32);
      lrow[i] = lrow[i] * fac + ls;
#pragma unroll
      for (int t = 0; t < 4; ++t) oacc[t][i] *= fac;
    }

    // stage P (wave-local) and accumulate O += P @ V
#pragma unroll
    for (int t = 0; t < 8; ++t)
#pragma unroll
      for (int i = 0; i < 8; ++i)
        pw[(mb + i) * LDK + t * 16 + nl] = (_Float16)sc[t][i];
    __builtin_amdgcn_wave_barrier();
#pragma unroll
    for (int t = 0; t < 4; ++t) {
      v8f c = oacc[t];
#pragma unroll
      for (int ks = 0; ks < 4; ++ks) {
        v16h pa = load_a(pw + ks * 32, LDK);
        v16h vb = load_b(vs + ks * 32 * LDW + t * 16, LDW);
        c = wmma_f16(pa, vb, c);
      }
      oacc[t] = c;
    }
  }

  // normalize by row sums and stage O (reuse pw)
#pragma unroll
  for (int t = 0; t < 4; ++t)
#pragma unroll
    for (int i = 0; i < 8; ++i)
      pw[(mb + i) * LDK + t * 16 + nl] = (_Float16)(oacc[t][i] / lrow[i]);
  __builtin_amdgcn_wave_barrier();

  // Z = O @ Wo^T + bo ; result = Xq + Z
  v16h oa0 = load_a(pw, LDK);
  v16h oa1 = load_a(pw + 32, LDK);
#pragma unroll
  for (int t = 0; t < 4; ++t) {
    v8f c = {};
    c = wmma_f16(oa0, load_b(wo + t * 16, LDW), c);
    c = wmma_f16(oa1, load_b(wo + 32 * LDW + t * 16, LDW), c);
    int f = t * 16 + nl;
    float bb = bo[f];
#pragma unroll
    for (int i = 0; i < 8; ++i) {
      int srow = qt * 128 + wave * 16 + mb + i;
      O[(size_t)srow * DD + f] = c[i] + bb + (float)Xq[(size_t)srow * DD + f];
    }
  }
}

// ---------------- Kernel 3: concat + final per-head LayerNorm over 128 ----------------

__global__ __launch_bounds__(256) void final_ln(const float* __restrict__ oIb,
                                                const float* __restrict__ oTb,
                                                const float* __restrict__ hg,
                                                const float* __restrict__ hb,
                                                float* __restrict__ out) {
  int wave = threadIdx.x >> 5, lane = threadIdx.x & 31;
  size_t row = (size_t)blockIdx.x * 8 + wave;   // (b*H + h)*S + s
  int h = (int)((row / SS) % HH);
  const float* xi = oIb + row * DD;
  const float* xt = oTb + row * DD;
  float v[4], sum = 0.f, sq = 0.f;
#pragma unroll
  for (int q = 0; q < 4; ++q) {
    int j = lane + q * 32;
    v[q] = (j < 64) ? xi[j] : xt[j - 64];
    sum += v[q]; sq += v[q] * v[q];
  }
  for (int off = 16; off >= 1; off >>= 1) { sum += __shfl_xor(sum, off, 32); sq += __shfl_xor(sq, off, 32); }
  float mu   = sum * (1.f / 128.f);
  float var  = sq * (1.f / 128.f) - mu * mu;
  float rstd = rsqrtf(var + 1e-5f);
  float* o = out + row * 128;
  const float* g  = hg + h * 128;
  const float* bb = hb + h * 128;
#pragma unroll
  for (int q = 0; q < 4; ++q) {
    int j = lane + q * 32;
    o[j] = (v[q] - mu) * rstd * g[j] + bb[j];
  }
}

// ---------------- Host launcher ----------------

extern "C" void kernel_launch(void* const* d_in, const int* in_sizes, int n_in,
                              void* d_out, int out_size, void* d_ws, size_t ws_size,
                              hipStream_t stream) {
  const float* imgF    = (const float*)d_in[0];
  const float* txtF    = (const float*)d_in[1];
  const float* img_w   = (const float*)d_in[2];
  const float* img_b   = (const float*)d_in[3];
  const float* img_g   = (const float*)d_in[4];
  const float* img_be  = (const float*)d_in[5];
  const float* txt_w   = (const float*)d_in[6];
  const float* txt_b   = (const float*)d_in[7];
  const float* txt_g   = (const float*)d_in[8];
  const float* txt_be  = (const float*)d_in[9];
  const float* i2t_iw  = (const float*)d_in[10];
  const float* i2t_ib  = (const float*)d_in[11];
  const float* i2t_ow  = (const float*)d_in[12];
  const float* i2t_ob  = (const float*)d_in[13];
  const float* t2i_iw  = (const float*)d_in[14];
  const float* t2i_ib  = (const float*)d_in[15];
  const float* t2i_ow  = (const float*)d_in[16];
  const float* t2i_ob  = (const float*)d_in[17];
  const float* hn_g    = (const float*)d_in[18];
  const float* hn_b    = (const float*)d_in[19];

  const size_t NE = (size_t)MM * HID;     // 8,388,608
  char* ws = (char*)d_ws;
  float*     yImg  = (float*)ws;
  float*     yTxt  = yImg + NE;
  _Float16*  ciH   = (_Float16*)(yTxt + NE);
  _Float16*  ctH   = ciH + NE;
  float*     oI    = (float*)(ctH + NE);
  float*     oT    = oI + NE;
  _Float16*  Xi16  = (_Float16*)(oT + NE);            // 8192*768
  _Float16*  Xt16  = Xi16 + (size_t)MM * 768;         // 8192*512
  _Float16*  WiT16 = Xt16 + (size_t)MM * 512;         // 768*1024
  _Float16*  WtT16 = WiT16 + (size_t)768 * HID;       // 512*1024
  _Float16*  wAtt  = WtT16 + (size_t)512 * HID;       // 2*16*4*64*64 = 524288

  dim3 blk(256);

  // ---- prep: f32 -> f16 (inputs, transposed weights, attention weights) ----
  cvt16<<<(MM * 768 / 4 + 255) / 256, blk, 0, stream>>>(imgF, Xi16, MM * 768 / 4);
  cvt16<<<(MM * 512 / 4 + 255) / 256, blk, 0, stream>>>(txtF, Xt16, MM * 512 / 4);
  cvtWT<<<(768 * HID + 255) / 256, blk, 0, stream>>>(img_w, WiT16, 768);
  cvtWT<<<(512 * HID + 255) / 256, blk, 0, stream>>>(txt_w, WtT16, 512);
  cvt_attn_w<<<(524288 + 255) / 256, blk, 0, stream>>>(i2t_iw, i2t_ow, t2i_iw, t2i_ow, wAtt);

  // ---- projections + LN ----
  const size_t smemG = (size_t)(2 * GEMM_BUF) * sizeof(_Float16);     // 37,888 B
  proj_gemm<<<dim3(MM / 128, HID / 128), blk, smemG, stream>>>(Xi16, WiT16, img_b, yImg, 768);
  proj_gemm<<<dim3(MM / 128, HID / 128), blk, smemG, stream>>>(Xt16, WtT16, txt_b, yTxt, 512);
  ln_l2_reshape<<<MM, blk, 0, stream>>>(yImg, img_g, img_be, ciH);
  ln_l2_reshape<<<MM, blk, 0, stream>>>(yTxt, txt_g, txt_be, ctH);

  // ---- fused cross attention ----
  const int LDW = 72, LDK = 136;
  const size_t smemA = (size_t)(4 * 64 * LDW + 3 * 128 * LDW + 64 * LDK +
                                128 * LDW + 8 * 16 * LDK) * sizeof(_Float16) +
                       4 * 64 * sizeof(float);   // 163,840 B
  (void)hipFuncSetAttribute((const void*)cross_attn_kernel,
                            hipFuncAttributeMaxDynamicSharedMemorySize, (int)smemA);
  cross_attn_kernel<<<dim3(8, HH, BB), blk, smemA, stream>>>(
      ciH, ctH, wAtt, i2t_ib, i2t_ob, t2i_ib, t2i_ob, oI, oT);

  // ---- concat + final LN ----
  final_ln<<<(BB * HH * SS) / 8, blk, 0, stream>>>(oI, oT, hn_g, hn_b, (float*)d_out);
}
